// CausalMultiHeadAttention_21036749816280
// MI455X (gfx1250) — compile-verified, matrix operand order checked by reading, not run on hardware
//
#include <hip/hip_runtime.h>
#include <hip/hip_bf16.h>

typedef __attribute__((ext_vector_type(16))) _Float16 v16h;
typedef __attribute__((ext_vector_type(8)))  _Float16 v8h;
typedef __attribute__((ext_vector_type(4)))  _Float16 v4h;
typedef __attribute__((ext_vector_type(8)))  float    v8f;

#define HEADS 16
#define TSEQ  2048
#define CDIM  1024
#define DK    64
#define NB    2

union Frag16 { v16h v; v8h h[2]; };

// ---------------------------------------------------------------------------
// CDNA5 async global->LDS copy (ASYNCcnt path), with sync fallback
// ---------------------------------------------------------------------------
#if defined(__has_builtin)
#if __has_builtin(__builtin_amdgcn_global_load_async_to_lds_b128)
#define USE_ASYNC_LDS 1
#endif
#endif

typedef __attribute__((__vector_size__(4 * sizeof(int)))) int avec4i;

__device__ __forceinline__ void cp16(_Float16* dst_lds, const _Float16* src_glb) {
#ifdef USE_ASYNC_LDS
  __builtin_amdgcn_global_load_async_to_lds_b128(
      (__attribute__((address_space(1))) avec4i*)src_glb,
      (__attribute__((address_space(3))) avec4i*)dst_lds, 0, 0);
#else
  *(uint4*)dst_lds = *(const uint4*)src_glb;
#endif
}

__device__ __forceinline__ void wait_async0() {
#ifdef USE_ASYNC_LDS
#if __has_builtin(__builtin_amdgcn_s_wait_asynccnt)
  __builtin_amdgcn_s_wait_asynccnt(0);
#else
  asm volatile("s_wait_asynccnt 0" ::: "memory");
#endif
#endif
}

// ---------------------------------------------------------------------------
// f32 -> f16 convert (optionally scaled); 4 elements per thread
// ---------------------------------------------------------------------------
__launch_bounds__(256)
__global__ void cvt_f32_to_f16(const float* __restrict__ in,
                               _Float16* __restrict__ out,
                               int n, float scale) {
  int i = (blockIdx.x * 256 + threadIdx.x) * 4;
  if (i < n) {
    float4 x = *(const float4*)(in + i);
    v4h y;
    y.x = (_Float16)(x.x * scale);
    y.y = (_Float16)(x.y * scale);
    y.z = (_Float16)(x.z * scale);
    y.w = (_Float16)(x.w * scale);
    *(v4h*)(out + i) = y;
  }
}

// ---------------------------------------------------------------------------
// Tiled WMMA GEMM:  Y[m,n] = sum_k A[m,k] * W[n,k]   (K = N = 1024)
// Block: 256 threads = 8 waves. Tile: M=128 (16 rows/wave), N=64, Kstep=64.
// Double-buffered LDS, async prefetch of tile i+1 overlaps WMMA on tile i.
// mode 0: write f16 [B,h,T,dk]     (Q/K projection)
// mode 1: write f16 [B,h,dk,T]     (V projection, transposed)
// mode 2: write f32 row-major      (output projection)
// ---------------------------------------------------------------------------
__launch_bounds__(256)
__global__ void gemm_wmma_f16(const _Float16* __restrict__ A,
                              const _Float16* __restrict__ W,
                              void* __restrict__ out, int mode) {
  const int K = CDIM, N = CDIM;
  __shared__ alignas(16) _Float16 As[2][128 * 72];   // 128 rows x 64 halfs
  __shared__ alignas(16) _Float16 Bs[2][64 * 72];

  const int tid  = threadIdx.x;
  const int wave = tid >> 5;
  const int lane = tid & 31;
  const int frow = lane & 15;          // fragment row (A) / col (B)
  const int hi   = lane >> 4;
  const int koff = hi * 8;             // ISA 16-bit A/B k-chunk base
  const int m0 = blockIdx.x * 128;
  const int n0 = blockIdx.y * 64;

  auto loadTiles = [&](int buf, int k0) {
    // A tile: 128 x 64 halfs = 1024 16B chunks, 4 per thread
    #pragma unroll
    for (int c = tid; c < 1024; c += 256) {
      int r = c >> 3, o = (c & 7) * 8;
      cp16(&As[buf][r * 72 + o], A + (size_t)(m0 + r) * K + k0 + o);
    }
    // B tile: 64 x 64 halfs = 512 chunks, 2 per thread
    #pragma unroll
    for (int c = tid; c < 512; c += 256) {
      int r = c >> 3, o = (c & 7) * 8;
      cp16(&Bs[buf][r * 72 + o], W + (size_t)(n0 + r) * K + k0 + o);
    }
  };

  v8f acc[4] = {};

  loadTiles(0, 0);
  wait_async0();
  __syncthreads();

  for (int k0 = 0; k0 < K; k0 += 64) {
    const int cur = (k0 >> 6) & 1;
    if (k0 + 64 < K) loadTiles(cur ^ 1, k0 + 64);   // prefetch next stage

    #pragma unroll
    for (int kc = 0; kc < 2; ++kc) {
      Frag16 a;
      a.h[0] = *(const v8h*)&As[cur][(wave * 16 + frow) * 72 + kc * 32 + koff];
      a.h[1] = *(const v8h*)&As[cur][(wave * 16 + frow) * 72 + kc * 32 + koff + 16];
      #pragma unroll
      for (int nt = 0; nt < 4; ++nt) {
        Frag16 b;
        b.h[0] = *(const v8h*)&Bs[cur][(nt * 16 + frow) * 72 + kc * 32 + koff];
        b.h[1] = *(const v8h*)&Bs[cur][(nt * 16 + frow) * 72 + kc * 32 + koff + 16];
        acc[nt] = __builtin_amdgcn_wmma_f32_16x16x32_f16(
            false, a.v, false, b.v, (short)0, acc[nt], false, false);
      }
    }

    wait_async0();
    __syncthreads();
  }

  // D layout: VGPR r -> M = r + 8*hi, N = frow
  #pragma unroll
  for (int nt = 0; nt < 4; ++nt) {
    #pragma unroll
    for (int r = 0; r < 8; ++r) {
      int m = m0 + wave * 16 + r + 8 * hi;
      int n = n0 + nt * 16 + frow;
      float val = acc[nt][r];
      if (mode == 2) {
        ((float*)out)[(size_t)m * N + n] = val;
      } else {
        int b = m >> 11, t = m & (TSEQ - 1);
        int head = n >> 6, d = n & 63;
        size_t off = (mode == 0)
            ? ((size_t)((b * HEADS + head) * TSEQ + t) * DK + d)
            : ((size_t)((b * HEADS + head) * DK + d) * TSEQ + t);
        ((_Float16*)out)[off] = (_Float16)val;
      }
    }
  }
}

// ---------------------------------------------------------------------------
// Flash attention, causal. Grid (T/128, B*HEADS), 256 threads = 8 waves.
// Each wave: one 16-row query block. Key tiles of 32, K/V^T double-buffered
// in LDS with async prefetch. Qh pre-scaled by 1/sqrt(dk).
// ---------------------------------------------------------------------------
__launch_bounds__(256)
__global__ void flash_attn_wmma(const _Float16* __restrict__ Qh,  // [B,H,T,DK]
                                const _Float16* __restrict__ Kh,  // [B,H,T,DK]
                                const _Float16* __restrict__ Vt,  // [B,H,DK,T]
                                _Float16* __restrict__ attnOut) { // [B*T, C]
  __shared__ alignas(16) _Float16 Ks[2][32 * 72];    // 32 keys x 64 d
  __shared__ alignas(16) _Float16 Vs[2][64 * 40];    // 64 d x 32 keys
  __shared__ alignas(16) _Float16 Ps[8][16 * 40];    // per-wave P transpose buf

  const int tid  = threadIdx.x;
  const int wave = tid >> 5;
  const int lane = tid & 31;
  const int frow = lane & 15;
  const int hi   = lane >> 4;
  const int koff = hi * 8;

  const int bh   = blockIdx.y;
  const int b    = bh / HEADS;
  const int head = bh % HEADS;
  const int q0   = blockIdx.x * 128 + wave * 16;

  const _Float16* Qbase = Qh + (size_t)bh * TSEQ * DK;
  const _Float16* Kbase = Kh + (size_t)bh * TSEQ * DK;
  const _Float16* Vbase = Vt + (size_t)bh * DK * TSEQ;

  auto loadKV = [&](int buf, int j0) {
    int kr = tid >> 3, ko = (tid & 7) * 8;   // K tile: 32x64 = 256 chunks
    cp16(&Ks[buf][kr * 72 + ko], Kbase + (size_t)(j0 + kr) * DK + ko);
    int d = tid >> 2, vo = (tid & 3) * 8;    // V^T tile: 64x32 = 256 chunks
    cp16(&Vs[buf][d * 40 + vo], Vbase + (size_t)d * TSEQ + j0 + vo);
  };

  // Q fragments for k=0..31 and k=32..63
  Frag16 qf[2];
  {
    const _Float16* qp = Qbase + (size_t)(q0 + frow) * DK;
    qf[0].h[0] = *(const v8h*)(qp + koff);
    qf[0].h[1] = *(const v8h*)(qp + koff + 16);
    qf[1].h[0] = *(const v8h*)(qp + 32 + koff);
    qf[1].h[1] = *(const v8h*)(qp + 32 + koff + 16);
  }

  v8f o[4] = {};
  float mrow[8], lsum[8];
  #pragma unroll
  for (int r = 0; r < 8; ++r) { mrow[r] = -3.0e38f; lsum[r] = 0.0f; }

  const int ntiles = (blockIdx.x + 1) * 4;   // causal: keys 0 .. wgMaxQ

  loadKV(0, 0);
  wait_async0();
  __syncthreads();

  for (int tile = 0; tile < ntiles; ++tile) {
    const int j0  = tile * 32;
    const int cur = tile & 1;
    if (tile + 1 < ntiles) loadKV(cur ^ 1, j0 + 32);   // prefetch next tile

    if (j0 <= q0 + 15) {   // wave-uniform: tile has unmasked keys for us
      // S = Q * K^T : two 16x16 score tiles, dk=64 -> 2 WMMAs each
      v8f s[2] = {};
      #pragma unroll
      for (int nt = 0; nt < 2; ++nt) {
        #pragma unroll
        for (int kc = 0; kc < 2; ++kc) {
          Frag16 kf;
          kf.h[0] = *(const v8h*)&Ks[cur][(nt * 16 + frow) * 72 + kc * 32 + koff];
          kf.h[1] = *(const v8h*)&Ks[cur][(nt * 16 + frow) * 72 + kc * 32 + koff + 16];
          s[nt] = __builtin_amdgcn_wmma_f32_16x16x32_f16(
              false, qf[kc].v, false, kf.v, (short)0, s[nt], false, false);
        }
      }

      // causal mask (matches reference's -1e9 fill)
      if (j0 + 31 > q0) {
        #pragma unroll
        for (int nt = 0; nt < 2; ++nt)
          #pragma unroll
          for (int r = 0; r < 8; ++r) {
            int key = j0 + nt * 16 + frow;
            int qq  = q0 + r + 8 * hi;
            if (key > qq) s[nt][r] = -1.0e9f;
          }
      }

      // online softmax (rows live per-VGPR across each 16-lane half)
      _Float16* P = &Ps[wave][0];
      #pragma unroll
      for (int r = 0; r < 8; ++r) {
        float sm = fmaxf(s[0][r], s[1][r]);
        sm = fmaxf(sm, __shfl_xor(sm, 1, 32));
        sm = fmaxf(sm, __shfl_xor(sm, 2, 32));
        sm = fmaxf(sm, __shfl_xor(sm, 4, 32));
        sm = fmaxf(sm, __shfl_xor(sm, 8, 32));
        float mn    = fmaxf(mrow[r], sm);
        float alpha = __expf(mrow[r] - mn);
        mrow[r] = mn;
        float p0 = __expf(s[0][r] - mn);
        float p1 = __expf(s[1][r] - mn);
        float ps = p0 + p1;
        ps += __shfl_xor(ps, 1, 32);
        ps += __shfl_xor(ps, 2, 32);
        ps += __shfl_xor(ps, 4, 32);
        ps += __shfl_xor(ps, 8, 32);
        lsum[r] = lsum[r] * alpha + ps;
        o[0][r] *= alpha; o[1][r] *= alpha; o[2][r] *= alpha; o[3][r] *= alpha;
        int prow = r + 8 * hi;                 // D-layout -> LDS row-major P
        P[prow * 40 + frow]      = (_Float16)p0;
        P[prow * 40 + 16 + frow] = (_Float16)p1;
      }

      // wave-private LDS transpose turnaround
      asm volatile("s_wait_dscnt 0" ::: "memory");

      Frag16 pf;   // P as A-operand, K=32 keys
      pf.h[0] = *(const v8h*)&P[frow * 40 + koff];
      pf.h[1] = *(const v8h*)&P[frow * 40 + koff + 16];

      // O += P * V  (4 d-subtiles of 16)
      #pragma unroll
      for (int dt = 0; dt < 4; ++dt) {
        Frag16 vf;
        vf.h[0] = *(const v8h*)&Vs[cur][(dt * 16 + frow) * 40 + koff];
        vf.h[1] = *(const v8h*)&Vs[cur][(dt * 16 + frow) * 40 + koff + 16];
        o[dt] = __builtin_amdgcn_wmma_f32_16x16x32_f16(
            false, pf.v, false, vf.v, (short)0, o[dt], false, false);
      }
    }

    wait_async0();     // our prefetch landed
    __syncthreads();   // everyone's landed; buffer swap is safe
  }

  // normalize + write f16 concat layout [B*T, C]
  #pragma unroll
  for (int r = 0; r < 8; ++r) {
    float inv = 1.0f / lsum[r];
    int qq = q0 + r + 8 * hi;
    size_t row = ((size_t)b * TSEQ + qq) * CDIM + head * DK;
    #pragma unroll
    for (int dt = 0; dt < 4; ++dt)
      attnOut[row + dt * 16 + frow] = (_Float16)(o[dt][r] * inv);
  }
}

// ---------------------------------------------------------------------------
extern "C" void kernel_launch(void* const* d_in, const int* in_sizes, int n_in,
                              void* d_out, int out_size, void* d_ws, size_t ws_size,
                              hipStream_t stream) {
  const float* q  = (const float*)d_in[0];
  const float* k  = (const float*)d_in[1];
  const float* v  = (const float*)d_in[2];
  const float* Wq = (const float*)d_in[3];
  const float* Wk = (const float*)d_in[4];
  const float* Wv = (const float*)d_in[5];
  const float* Wo = (const float*)d_in[6];

  char* ws = (char*)d_ws;
  const size_t MB = 1u << 20;
  _Float16* Wq16 = (_Float16*)(ws + 0 * MB);   // 2 MB each
  _Float16* Wk16 = (_Float16*)(ws + 2 * MB);
  _Float16* Wv16 = (_Float16*)(ws + 4 * MB);
  _Float16* Wo16 = (_Float16*)(ws + 6 * MB);
  _Float16* Xq16 = (_Float16*)(ws + 8 * MB);   // 8 MB each
  _Float16* Xk16 = (_Float16*)(ws + 16 * MB);
  _Float16* Xv16 = (_Float16*)(ws + 24 * MB);
  _Float16* Qh   = (_Float16*)(ws + 32 * MB);
  _Float16* Kh   = (_Float16*)(ws + 40 * MB);
  _Float16* Vt   = (_Float16*)(ws + 48 * MB);
  _Float16* attn = (_Float16*)(ws + 8 * MB);   // reuse Xq16 (retired by then)

  const int nW = CDIM * CDIM;                  // 1M
  const int nX = NB * TSEQ * CDIM;             // 4M
  const float qscale = 0.125f;                 // 1/sqrt(dk), folded into Wq

  cvt_f32_to_f16<<<nW / 1024, 256, 0, stream>>>(Wq, Wq16, nW, qscale);
  cvt_f32_to_f16<<<nW / 1024, 256, 0, stream>>>(Wk, Wk16, nW, 1.0f);
  cvt_f32_to_f16<<<nW / 1024, 256, 0, stream>>>(Wv, Wv16, nW, 1.0f);
  cvt_f32_to_f16<<<nW / 1024, 256, 0, stream>>>(Wo, Wo16, nW, 1.0f);
  cvt_f32_to_f16<<<nX / 1024, 256, 0, stream>>>(q, Xq16, nX, 1.0f);
  cvt_f32_to_f16<<<nX / 1024, 256, 0, stream>>>(k, Xk16, nX, 1.0f);
  cvt_f32_to_f16<<<nX / 1024, 256, 0, stream>>>(v, Xv16, nX, 1.0f);

  dim3 ggrid(32, 16);   // M/128, N/64
  gemm_wmma_f16<<<ggrid, 256, 0, stream>>>(Xq16, Wq16, Qh, 0);
  gemm_wmma_f16<<<ggrid, 256, 0, stream>>>(Xk16, Wk16, Kh, 0);
  gemm_wmma_f16<<<ggrid, 256, 0, stream>>>(Xv16, Wv16, Vt, 1);

  flash_attn_wmma<<<dim3(16, NB * HEADS), 256, 0, stream>>>(Qh, Kh, Vt, attn);

  gemm_wmma_f16<<<ggrid, 256, 0, stream>>>(attn, Wo16, (float*)d_out, 2);
}
